// _EmformerLayer_32950989094998
// MI455X (gfx1250) — compile-verified, matrix operand order checked by reading, not run on hardware
//
#include <hip/hip_runtime.h>
#include <hip/hip_bf16.h>

// ---------------------------------------------------------------------------
// Emformer layer for MI455X (gfx1250, wave32, WMMA).
// All GEMMs use v_wmma_f32_16x16x32_bf16 (bf16 in, f32 accumulate).
// ---------------------------------------------------------------------------

typedef __bf16 bf16_t;
typedef bf16_t v16bf __attribute__((ext_vector_type(16)));
typedef bf16_t v8bf  __attribute__((ext_vector_type(8)));
typedef float  v8f   __attribute__((ext_vector_type(8)));

#define NEG_INF (-1.0e8f)

// Problem constants
#define DDIM   1024
#define HEADS  16
#define HD     64
#define FFN    4096
#define BATCH  64
#define ULEN   128
#define RLEN   32
#define MMEM   4
#define LCTX   256
#define SEG    128
#define QLEN   161        // R + U + 1 summary
#define KLEN   420        // M + R + L + U
#define KPAD   448        // KLEN padded to multiple of 32
#define PSTR   (KPAD + 4) // LDS row stride for scores

// Row counts (time-major flatten, row = t*BATCH + b)
#define LN_ROWS   10240   // (R+U)*B
#define Q_ROWS    10304   // QLEN*B
#define KV_ROWS   10496   // (M+R+U)*B
#define X_ROWS    10560   // mems(256) + ln(10240) + summary(64)

// ---------------- workspace layout (bytes) ---------------------------------
// live early:  X, Q, K, VT, ATTN    live late: H, FFOUT (reuse early region)
#define WS_X      ((size_t)0)                        // 10560*1024*2  = 21,626,880
#define WS_Q      ((size_t)21626880)                 // 10304*1024*2  = 21,102,592
#define WS_K      ((size_t)42729472)                 // 448*64*1024*2 = 58,720,256
#define WS_VT     ((size_t)101449728)                // 65536*448*2   = 58,720,256
#define WS_ATTN   ((size_t)160169984)                // 10304*1024*2  = 21,102,592
#define WS_RES    ((size_t)181272576)                // 10240*1024*4  = 41,943,040
#define WS_FFIN   ((size_t)223215616)                // 10240*1024*2  = 20,971,520
#define WS_H      ((size_t)0)                        // 10240*4096*2  = 83,886,080 (reuse)
#define WS_FFOUT  ((size_t)83886080)                 // 10240*1024*4  = 41,943,040 (reuse)

// ---------------- d_out layout (floats) ------------------------------------
#define O_UTT     ((size_t)0)
#define O_RC      ((size_t)8388608)
#define O_MEMS    ((size_t)10485760)
#define O_NEWMEMS ((size_t)10551296)
#define O_KEY     ((size_t)10813440)
#define O_VAL     ((size_t)27590656)

// ---------------------------------------------------------------------------
static __device__ inline v16bf lds_ld16(const bf16_t* p) {
  v8bf lo = *(const v8bf*)p;
  v8bf hi = *(const v8bf*)(p + 8);
  return __builtin_shufflevector(lo, hi, 0,1,2,3,4,5,6,7,8,9,10,11,12,13,14,15);
}
static __device__ inline void lds_st16(bf16_t* p, v16bf v) {
  *(v8bf*)p       = __builtin_shufflevector(v, v, 0,1,2,3,4,5,6,7);
  *(v8bf*)(p + 8) = __builtin_shufflevector(v, v, 8,9,10,11,12,13,14,15);
}
static __device__ inline v8f wmma_bf16(v16bf a, v16bf b, v8f c) {
  return __builtin_amdgcn_wmma_f32_16x16x32_bf16(false, a, false, b, (short)0, c,
                                                 false, false);
}

// ---------------------------------------------------------------------------
// state_mems (f32) -> X rows [0,256) as bf16
__global__ __launch_bounds__(256) void k_mems_to_x(const float* __restrict__ smems,
                                                   bf16_t* __restrict__ X) {
  int row = blockIdx.x;                       // 0..255
  const float4 v = ((const float4*)(smems + (size_t)row * DDIM))[threadIdx.x];
  bf16_t* d = X + (size_t)row * DDIM + threadIdx.x * 4;
  d[0] = (bf16_t)v.x; d[1] = (bf16_t)v.y; d[2] = (bf16_t)v.z; d[3] = (bf16_t)v.w;
}

// LayerNorm over concat(rc, utt) row -> X rows [256, 256+10240) as bf16
__global__ __launch_bounds__(256) void k_ln_in(const float* __restrict__ rc,
                                               const float* __restrict__ utt,
                                               const float* __restrict__ g,
                                               const float* __restrict__ bta,
                                               bf16_t* __restrict__ X) {
  int row = blockIdx.x;                       // 0..10239
  int t = row >> 6, b = row & 63;
  const float* src = (t < RLEN) ? rc  + ((size_t)(t * BATCH + b)) * DDIM
                                : utt + ((size_t)((t - RLEN) * BATCH + b)) * DDIM;
  float4 x = ((const float4*)src)[threadIdx.x];
  float s  = x.x + x.y + x.z + x.w;
  float ss = x.x * x.x + x.y * x.y + x.z * x.z + x.w * x.w;
  for (int o = 16; o > 0; o >>= 1) { s += __shfl_down(s, o); ss += __shfl_down(ss, o); }
  __shared__ float sm[8], sm2[8], st[2];
  int lane = threadIdx.x & 31, w = threadIdx.x >> 5;
  if (lane == 0) { sm[w] = s; sm2[w] = ss; }
  __syncthreads();
  if (threadIdx.x == 0) {
    float S = 0.f, S2 = 0.f;
    for (int i = 0; i < 8; ++i) { S += sm[i]; S2 += sm2[i]; }
    float mu = S * (1.f / DDIM);
    st[0] = mu; st[1] = rsqrtf(S2 * (1.f / DDIM) - mu * mu + 1e-5f);
  }
  __syncthreads();
  float mu = st[0], rs = st[1];
  int d0 = threadIdx.x * 4;
  bf16_t* dst = X + (size_t)(256 + row) * DDIM + d0;
  dst[0] = (bf16_t)((x.x - mu) * rs * g[d0 + 0] + bta[d0 + 0]);
  dst[1] = (bf16_t)((x.y - mu) * rs * g[d0 + 1] + bta[d0 + 1]);
  dst[2] = (bf16_t)((x.z - mu) * rs * g[d0 + 2] + bta[d0 + 2]);
  dst[3] = (bf16_t)((x.w - mu) * rs * g[d0 + 3] + bta[d0 + 3]);
}

// summary = mean over 128 utterance time steps -> X rows [10496, 10560)
__global__ __launch_bounds__(256) void k_summary(bf16_t* __restrict__ X) {
  int b = blockIdx.x;
  int d0 = threadIdx.x * 4;
  float a0 = 0.f, a1 = 0.f, a2 = 0.f, a3 = 0.f;
  for (int t = 0; t < SEG; ++t) {
    const bf16_t* p = X + (size_t)(256 + (RLEN + t) * BATCH + b) * DDIM + d0;
    a0 += (float)p[0]; a1 += (float)p[1]; a2 += (float)p[2]; a3 += (float)p[3];
  }
  bf16_t* q = X + (size_t)(256 + LN_ROWS + b) * DDIM + d0;
  const float inv = 1.f / SEG;
  q[0] = (bf16_t)(a0 * inv); q[1] = (bf16_t)(a1 * inv);
  q[2] = (bf16_t)(a2 * inv); q[3] = (bf16_t)(a3 * inv);
}

// state_lc_key/val -> K rows [36,292) and VT, plus pad rows [420,448) = 0,
// plus first halves of new_lc_key/new_lc_val outputs (f32).
__global__ __launch_bounds__(256) void k_lc_copy(const float* __restrict__ skey,
                                                 const float* __restrict__ sval,
                                                 bf16_t* __restrict__ K,
                                                 bf16_t* __restrict__ VT,
                                                 float* __restrict__ okey,
                                                 float* __restrict__ oval) {
  int idx = blockIdx.x;                       // 0..283
  int b = blockIdx.y;
  int kr = (idx < LCTX) ? (MMEM + RLEN + idx) : (KLEN + (idx - LCTX));
  int d0 = threadIdx.x * 4;
  if (kr < MMEM + RLEN + LCTX) {
    int i = kr - (MMEM + RLEN);
    const float* kp = skey + ((size_t)(i * BATCH + b)) * DDIM + d0;
    const float* vp = sval + ((size_t)(i * BATCH + b)) * DDIM + d0;
#pragma unroll
    for (int e = 0; e < 4; ++e) {
      int d = d0 + e;
      float kv = kp[e], vv = vp[e];
      K[((size_t)(kr * BATCH + b)) * DDIM + d] = (bf16_t)kv;
      VT[((size_t)((b * HEADS + (d >> 6)) * HD + (d & 63))) * KPAD + kr] = (bf16_t)vv;
      if (i >= 128) {
        okey[((size_t)((i - 128) * BATCH + b)) * DDIM + d] = kv;
        oval[((size_t)((i - 128) * BATCH + b)) * DDIM + d] = vv;
      }
    }
  } else {
#pragma unroll
    for (int e = 0; e < 4; ++e) {
      int d = d0 + e;
      K[((size_t)(kr * BATCH + b)) * DDIM + d] = (bf16_t)0.f;
      VT[((size_t)((b * HEADS + (d >> 6)) * HD + (d & 63))) * KPAD + kr] = (bf16_t)0.f;
    }
  }
}

// new_mems = [state_mems[1:4], mems_in]
__global__ __launch_bounds__(256) void k_new_mems(const float* __restrict__ smems,
                                                  const float* __restrict__ memsi,
                                                  float* __restrict__ dst) {
  int row = blockIdx.x;                       // 0..255
  int i = row >> 6, b = row & 63;
  const float* src = (i < 3) ? smems + ((size_t)((i + 1) * BATCH + b)) * DDIM
                             : memsi + (size_t)b * DDIM;
  ((float4*)(dst + (size_t)row * DDIM))[threadIdx.x] = ((const float4*)src)[threadIdx.x];
}

// ---------------------------------------------------------------------------
// Tiled WMMA GEMM:  C[M,N] = A(bf16, MxK) * W(f32, NxK)^T + bias, fused epilogue.
// Block 256 threads = 8 waves (2x4), block tile 64x128, wave tile 32x32, BK=64.
#define BM 64
#define BN 128
#define BK 64
#define BKP (BK + 16)

struct GemmArgs {
  const bf16_t* A;
  const float*  W;
  const float*  bias;
  int Mrows, Kdim;
  bf16_t* ob0; bf16_t* ob1;
  float*  of0; float*  of1;
  const float* r0; const float* r1;
};

template <int MODE>
__global__ __launch_bounds__(256) void k_gemm(GemmArgs ga) {
  __shared__ bf16_t As[BM][BKP];
  __shared__ bf16_t Bs[BN][BKP];
  const int tid  = threadIdx.x;
  const int lane = tid & 31;
  const int wm   = (tid >> 5) >> 2;   // 0..1
  const int wn   = (tid >> 5) & 3;    // 0..3
  const int m0   = blockIdx.x * BM;
  const int n0   = blockIdx.y * BN;

  v8f acc[2][2] = {};

  const int arow = tid & 63;
  const int acol = (tid >> 6) * 16;
  const int brow = tid & 127;
  const int bcol = (tid >> 7) * 32;
  int gar = m0 + arow; if (gar >= ga.Mrows) gar = ga.Mrows - 1;
  const bf16_t* aptr = ga.A + (size_t)gar * ga.Kdim + acol;
  const float*  wptr = ga.W + (size_t)(n0 + brow) * ga.Kdim + bcol;

  const int ksteps = ga.Kdim / BK;
  for (int kt = 0; kt < ksteps; ++kt) {
    const int k0 = kt * BK;
    v16bf av = *(const v16bf*)(aptr + k0);
    v16bf bv0, bv1;
#pragma unroll
    for (int q4 = 0; q4 < 4; ++q4) {
      float4 w = ((const float4*)(wptr + k0))[q4];
      bv0[q4 * 4 + 0] = (bf16_t)w.x; bv0[q4 * 4 + 1] = (bf16_t)w.y;
      bv0[q4 * 4 + 2] = (bf16_t)w.z; bv0[q4 * 4 + 3] = (bf16_t)w.w;
    }
#pragma unroll
    for (int q4 = 0; q4 < 4; ++q4) {
      float4 w = ((const float4*)(wptr + k0 + 16))[q4];
      bv1[q4 * 4 + 0] = (bf16_t)w.x; bv1[q4 * 4 + 1] = (bf16_t)w.y;
      bv1[q4 * 4 + 2] = (bf16_t)w.z; bv1[q4 * 4 + 3] = (bf16_t)w.w;
    }
    if (kt + 1 < ksteps) {                    // CDNA5 global_prefetch path
      __builtin_prefetch(aptr + k0 + BK, 0, 0);
      __builtin_prefetch(wptr + k0 + BK, 0, 0);
    }
    __syncthreads();
    lds_st16(&As[arow][acol], av);
    lds_st16(&Bs[brow][bcol], bv0);
    lds_st16(&Bs[brow][bcol + 16], bv1);
    __syncthreads();
#pragma unroll
    for (int kk = 0; kk < 2; ++kk) {
      const int ko = kk * 32 + (lane >> 4) * 16;
      v16bf a0 = lds_ld16(&As[wm * 32      + (lane & 15)][ko]);
      v16bf a1 = lds_ld16(&As[wm * 32 + 16 + (lane & 15)][ko]);
      v16bf b0 = lds_ld16(&Bs[wn * 32      + (lane & 15)][ko]);
      v16bf b1 = lds_ld16(&Bs[wn * 32 + 16 + (lane & 15)][ko]);
      acc[0][0] = wmma_bf16(a0, b0, acc[0][0]);
      acc[0][1] = wmma_bf16(a0, b1, acc[0][1]);
      acc[1][0] = wmma_bf16(a1, b0, acc[1][0]);
      acc[1][1] = wmma_bf16(a1, b1, acc[1][1]);
    }
  }

  // Epilogue. C layout: vgpr j -> M = j + 8*(lane>=16), N = lane&15.
#pragma unroll
  for (int mi = 0; mi < 2; ++mi)
#pragma unroll
    for (int ni = 0; ni < 2; ++ni)
#pragma unroll
      for (int j = 0; j < 8; ++j) {
        int m = m0 + wm * 32 + mi * 16 + j + 8 * (lane >> 4);
        int n = n0 + wn * 32 + ni * 16 + (lane & 15);
        if (m >= ga.Mrows) continue;
        float val = acc[mi][ni][j] + ga.bias[n];
        if constexpr (MODE == 0) {            // Q projection, pre-scale hd^-0.5
          ga.ob0[(size_t)m * DDIM + n] = (bf16_t)(val * 0.125f);
        } else if constexpr (MODE == 1) {     // KV projection: scatter K / VT
          int t = m >> 6, b = m & 63;
          int kr = (t < MMEM + RLEN) ? t : t + LCTX;
          if (n < DDIM) {
            ga.ob0[((size_t)(kr * BATCH + b)) * DDIM + n] = (bf16_t)val;
            if (t >= MMEM + RLEN)
              ga.of0[((size_t)((128 + t - (MMEM + RLEN)) * BATCH + b)) * DDIM + n] = val;
          } else {
            int dn = n - DDIM;
            ga.ob1[((size_t)((b * HEADS + (dn >> 6)) * HD + (dn & 63))) * KPAD + kr] =
                (bf16_t)val;
            if (t >= MMEM + RLEN)
              ga.of1[((size_t)((128 + t - (MMEM + RLEN)) * BATCH + b)) * DDIM + dn] = val;
          }
        } else if constexpr (MODE == 2) {     // out proj: residual / mems clip
          if (m < LN_ROWS) {
            int t = m >> 6, b = m & 63;
            float orig = (t < RLEN)
                ? ga.r0[((size_t)(t * BATCH + b)) * DDIM + n]
                : ga.r1[((size_t)((t - RLEN) * BATCH + b)) * DDIM + n];
            ga.of0[(size_t)m * DDIM + n] = val + orig;
          } else {
            int b = m - LN_ROWS;
            ga.of1[(size_t)b * DDIM + n] = fminf(fmaxf(val, -10.f), 10.f);
          }
        } else if constexpr (MODE == 3) {     // FFN1 + ReLU
          ga.ob0[(size_t)m * FFN + n] = (bf16_t)fmaxf(val, 0.f);
        } else {                              // FFN2 -> f32
          ga.of0[(size_t)m * DDIM + n] = val;
        }
      }
}

// ---------------------------------------------------------------------------
// Attention: one wave per (head n = b*16+h, q-tile of 16 rows).
__global__ __launch_bounds__(32) void k_attn(const bf16_t* __restrict__ qb,
                                             const bf16_t* __restrict__ kb,
                                             const bf16_t* __restrict__ vT,
                                             bf16_t* __restrict__ attnb,
                                             const int* __restrict__ past_p) {
  __shared__ float P[16][PSTR];
  const int n = blockIdx.x;                   // 0..1023
  const int b = n >> 4, h = n & 15;
  const int qt = blockIdx.y;                  // 0..10
  const int lane = threadIdx.x;
  const int past = *past_p;
  const int m_kv = past < LCTX ? past : LCTX;
  const int m_m  = (past / SEG) < MMEM ? (past / SEG) : MMEM;

  int qrow = qt * 16 + (lane & 15);
  if (qrow > QLEN - 1) qrow = QLEN - 1;
  const bf16_t* qp = qb + ((size_t)(qrow * BATCH + b)) * DDIM + h * HD;
  const int klo = (lane >> 4) * 16;
  v16bf aq0 = *(const v16bf*)(qp + klo);
  v16bf aq1 = *(const v16bf*)(qp + 32 + klo);

  // S = Q K^T (scaled Q already), masked, into LDS
  for (int kt = 0; kt < KPAD / 16; ++kt) {
    const int kr = kt * 16 + (lane & 15);
    const bf16_t* kp = kb + ((size_t)(kr * BATCH + b)) * DDIM + h * HD;
    v16bf bk0 = *(const v16bf*)(kp + klo);
    v16bf bk1 = *(const v16bf*)(kp + 32 + klo);
    v8f s = {};
    s = wmma_bf16(aq0, bk0, s);
    s = wmma_bf16(aq1, bk1, s);
    const int col = kt * 16 + (lane & 15);
#pragma unroll
    for (int j = 0; j < 8; ++j) {
      const int m = j + 8 * (lane >> 4);
      const int grow = qt * 16 + m;
      float val = s[j];
      bool mask = (col >= KLEN) ||
                  (col < MMEM - m_m) ||
                  (col >= MMEM + RLEN && col < MMEM + RLEN + LCTX - m_kv) ||
                  (grow == QLEN - 1 && col < MMEM);
      P[m][col] = mask ? NEG_INF : val;
    }
  }
  __syncthreads();

  // softmax (lanes 0..15 own one row each)
  if (lane < 16) {
    float mx = NEG_INF;
    for (int c = 0; c < KPAD; ++c) mx = fmaxf(mx, P[lane][c]);
    float sum = 0.f;
    for (int c = 0; c < KPAD; ++c) {
      float e = __expf(P[lane][c] - mx);
      P[lane][c] = e;
      sum += e;
    }
    float inv = 1.f / sum;
    for (int c = 0; c < KPAD; ++c) P[lane][c] *= inv;
  }
  __syncthreads();

  // O = P V  (A = P tile from LDS, B = VT rows, krow-contiguous)
  v8f o[4] = {};
  for (int ks = 0; ks < KPAD / 32; ++ks) {
    const int c0 = ks * 32 + klo;
    v16bf pa;
#pragma unroll
    for (int e = 0; e < 16; ++e) pa[e] = (bf16_t)P[lane & 15][c0 + e];
#pragma unroll
    for (int dt = 0; dt < 4; ++dt) {
      const bf16_t* vp = vT + ((size_t)(n * HD + dt * 16 + (lane & 15))) * KPAD + c0;
      v16bf vb = *(const v16bf*)vp;
      o[dt] = wmma_bf16(pa, vb, o[dt]);
    }
  }
#pragma unroll
  for (int dt = 0; dt < 4; ++dt)
#pragma unroll
    for (int j = 0; j < 8; ++j) {
      const int m = j + 8 * (lane >> 4);
      const int grow = qt * 16 + m;
      if (grow < QLEN)
        attnb[((size_t)(grow * BATCH + b)) * DDIM + h * HD + dt * 16 + (lane & 15)] =
            (bf16_t)o[dt][j];
    }
}

// ---------------------------------------------------------------------------
// LN(res) -> bf16 (FFN input)
__global__ __launch_bounds__(256) void k_ln_ff(const float* __restrict__ res,
                                               const float* __restrict__ g,
                                               const float* __restrict__ bta,
                                               bf16_t* __restrict__ dst) {
  int row = blockIdx.x;
  float4 x = ((const float4*)(res + (size_t)row * DDIM))[threadIdx.x];
  float s  = x.x + x.y + x.z + x.w;
  float ss = x.x * x.x + x.y * x.y + x.z * x.z + x.w * x.w;
  for (int o = 16; o > 0; o >>= 1) { s += __shfl_down(s, o); ss += __shfl_down(ss, o); }
  __shared__ float sm[8], sm2[8], st[2];
  int lane = threadIdx.x & 31, w = threadIdx.x >> 5;
  if (lane == 0) { sm[w] = s; sm2[w] = ss; }
  __syncthreads();
  if (threadIdx.x == 0) {
    float S = 0.f, S2 = 0.f;
    for (int i = 0; i < 8; ++i) { S += sm[i]; S2 += sm2[i]; }
    float mu = S * (1.f / DDIM);
    st[0] = mu; st[1] = rsqrtf(S2 * (1.f / DDIM) - mu * mu + 1e-5f);
  }
  __syncthreads();
  float mu = st[0], rs = st[1];
  int d0 = threadIdx.x * 4;
  bf16_t* d = dst + (size_t)row * DDIM + d0;
  d[0] = (bf16_t)((x.x - mu) * rs * g[d0 + 0] + bta[d0 + 0]);
  d[1] = (bf16_t)((x.y - mu) * rs * g[d0 + 1] + bta[d0 + 1]);
  d[2] = (bf16_t)((x.z - mu) * rs * g[d0 + 2] + bta[d0 + 2]);
  d[3] = (bf16_t)((x.w - mu) * rs * g[d0 + 3] + bta[d0 + 3]);
}

// final: LN(res + ffout) -> d_out utterance / right_context (f32)
__global__ __launch_bounds__(256) void k_final(const float* __restrict__ res,
                                               const float* __restrict__ ffo,
                                               const float* __restrict__ g,
                                               const float* __restrict__ bta,
                                               float* __restrict__ out_utt,
                                               float* __restrict__ out_rc) {
  int row = blockIdx.x;
  float4 a = ((const float4*)(res + (size_t)row * DDIM))[threadIdx.x];
  float4 f = ((const float4*)(ffo + (size_t)row * DDIM))[threadIdx.x];
  float4 x = make_float4(a.x + f.x, a.y + f.y, a.z + f.z, a.w + f.w);
  float s  = x.x + x.y + x.z + x.w;
  float ss = x.x * x.x + x.y * x.y + x.z * x.z + x.w * x.w;
  for (int o = 16; o > 0; o >>= 1) { s += __shfl_down(s, o); ss += __shfl_down(ss, o); }
  __shared__ float sm[8], sm2[8], st[2];
  int lane = threadIdx.x & 31, w = threadIdx.x >> 5;
  if (lane == 0) { sm[w] = s; sm2[w] = ss; }
  __syncthreads();
  if (threadIdx.x == 0) {
    float S = 0.f, S2 = 0.f;
    for (int i = 0; i < 8; ++i) { S += sm[i]; S2 += sm2[i]; }
    float mu = S * (1.f / DDIM);
    st[0] = mu; st[1] = rsqrtf(S2 * (1.f / DDIM) - mu * mu + 1e-5f);
  }
  __syncthreads();
  float mu = st[0], rs = st[1];
  int t = row >> 6, b = row & 63;
  float* dst = (t < RLEN) ? out_rc  + ((size_t)(t * BATCH + b)) * DDIM
                          : out_utt + ((size_t)((t - RLEN) * BATCH + b)) * DDIM;
  int d0 = threadIdx.x * 4;
  dst[d0 + 0] = (x.x - mu) * rs * g[d0 + 0] + bta[d0 + 0];
  dst[d0 + 1] = (x.y - mu) * rs * g[d0 + 1] + bta[d0 + 1];
  dst[d0 + 2] = (x.z - mu) * rs * g[d0 + 2] + bta[d0 + 2];
  dst[d0 + 3] = (x.w - mu) * rs * g[d0 + 3] + bta[d0 + 3];
}

// ---------------------------------------------------------------------------
extern "C" void kernel_launch(void* const* d_in, const int* in_sizes, int n_in,
                              void* d_out_, int out_size, void* d_ws, size_t ws_size,
                              hipStream_t stream) {
  const float* utt    = (const float*)d_in[0];
  const float* rc     = (const float*)d_in[1];
  const float* memsi  = (const float*)d_in[2];
  const float* smems  = (const float*)d_in[3];
  const float* skey   = (const float*)d_in[4];
  const float* sval   = (const float*)d_in[5];
  const float* W_kv   = (const float*)d_in[6];
  const float* b_kv   = (const float*)d_in[7];
  const float* W_q    = (const float*)d_in[8];
  const float* b_q    = (const float*)d_in[9];
  const float* W_out  = (const float*)d_in[10];
  const float* b_out  = (const float*)d_in[11];
  const float* ln_in_g = (const float*)d_in[12];
  const float* ln_in_b = (const float*)d_in[13];
  const float* ln_ff_g = (const float*)d_in[14];
  const float* ln_ff_b = (const float*)d_in[15];
  const float* W_ff1  = (const float*)d_in[16];
  const float* b_ff1  = (const float*)d_in[17];
  const float* W_ff2  = (const float*)d_in[18];
  const float* b_ff2  = (const float*)d_in[19];
  const float* ln_out_g = (const float*)d_in[20];
  const float* ln_out_b = (const float*)d_in[21];
  const int*   past   = (const int*)d_in[22];

  char* ws = (char*)d_ws;
  bf16_t* X    = (bf16_t*)(ws + WS_X);
  bf16_t* Qb   = (bf16_t*)(ws + WS_Q);
  bf16_t* Kb   = (bf16_t*)(ws + WS_K);
  bf16_t* VTb  = (bf16_t*)(ws + WS_VT);
  bf16_t* ATTN = (bf16_t*)(ws + WS_ATTN);
  float*  RES  = (float*)(ws + WS_RES);
  bf16_t* FFIN = (bf16_t*)(ws + WS_FFIN);
  bf16_t* Hb   = (bf16_t*)(ws + WS_H);
  float*  FFO  = (float*)(ws + WS_FFOUT);

  float* out = (float*)d_out_;

  // ---- stage 1: LN + buffer assembly + state copies -----------------------
  k_mems_to_x<<<256, 256, 0, stream>>>(smems, X);
  k_ln_in<<<LN_ROWS, 256, 0, stream>>>(rc, utt, ln_in_g, ln_in_b, X);
  k_summary<<<BATCH, 256, 0, stream>>>(X);
  k_lc_copy<<<dim3(284, 64), 256, 0, stream>>>(skey, sval, Kb, VTb,
                                               out + O_KEY, out + O_VAL);
  k_new_mems<<<256, 256, 0, stream>>>(smems, memsi, out + O_NEWMEMS);

  // ---- stage 2: projections (WMMA GEMMs) ----------------------------------
  {
    GemmArgs ga = {};
    ga.A = X + (size_t)256 * DDIM; ga.W = W_q; ga.bias = b_q;
    ga.Mrows = Q_ROWS; ga.Kdim = DDIM; ga.ob0 = Qb;
    k_gemm<0><<<dim3(Q_ROWS / BM, DDIM / BN), 256, 0, stream>>>(ga);
  }
  {
    GemmArgs ga = {};
    ga.A = X; ga.W = W_kv; ga.bias = b_kv;
    ga.Mrows = KV_ROWS; ga.Kdim = DDIM;
    ga.ob0 = Kb; ga.ob1 = VTb; ga.of0 = out + O_KEY; ga.of1 = out + O_VAL;
    k_gemm<1><<<dim3(KV_ROWS / BM, 2 * DDIM / BN), 256, 0, stream>>>(ga);
  }

  // ---- stage 3: attention -------------------------------------------------
  k_attn<<<dim3(BATCH * HEADS, (QLEN + 15) / 16), 32, 0, stream>>>(Qb, Kb, VTb, ATTN,
                                                                   past);

  // ---- stage 4: out projection + residual / mems --------------------------
  {
    GemmArgs ga = {};
    ga.A = ATTN; ga.W = W_out; ga.bias = b_out;
    ga.Mrows = Q_ROWS; ga.Kdim = DDIM;
    ga.of0 = RES; ga.of1 = out + O_MEMS; ga.r0 = rc; ga.r1 = utt;
    k_gemm<2><<<dim3(Q_ROWS / BM, DDIM / BN), 256, 0, stream>>>(ga);
  }

  // ---- stage 5: FFN -------------------------------------------------------
  k_ln_ff<<<LN_ROWS, 256, 0, stream>>>(RES, ln_ff_g, ln_ff_b, FFIN);
  {
    GemmArgs ga = {};
    ga.A = FFIN; ga.W = W_ff1; ga.bias = b_ff1;
    ga.Mrows = LN_ROWS; ga.Kdim = DDIM; ga.ob0 = Hb;
    k_gemm<3><<<dim3(LN_ROWS / BM, FFN / BN), 256, 0, stream>>>(ga);
  }
  {
    GemmArgs ga = {};
    ga.A = Hb; ga.W = W_ff2; ga.bias = b_ff2;
    ga.Mrows = LN_ROWS; ga.Kdim = FFN; ga.of0 = FFO;
    k_gemm<4><<<dim3(LN_ROWS / BM, DDIM / BN), 256, 0, stream>>>(ga);
  }

  // ---- stage 6: final LN -> outputs ---------------------------------------
  k_final<<<LN_ROWS, 256, 0, stream>>>(RES, FFO, ln_out_g, ln_out_b,
                                       out + O_UTT, out + O_RC);
}
